// SETransformerLayer_44212393345041
// MI455X (gfx1250) — compile-verified, matrix operand order checked by reading, not run on hardware
//
#include <hip/hip_runtime.h>
#include <hip/hip_bf16.h>
#include <math.h>

// SE-Transformer layer, fused for gfx1250 (wave32, WMMA).
// Pass 0: pre-swizzle W1/W2 B-fragments to f16 packed layout.
// Pass 1: node q precompute. Pass 2: per-edge fused radial-MLP (WMMA) + fctp +
// score + expv/z accumulation. Pass 3: alpha normalization + scatter.

typedef __attribute__((ext_vector_type(16))) _Float16 v16h;
typedef __attribute__((ext_vector_type(8)))  _Float16 v8h;
typedef __attribute__((ext_vector_type(8)))  float    v8f;

#define DCH   40     // M0 + 3*M1
#define WNUMC 576
#define NT2   36     // 576/16 N-tiles

__device__ __forceinline__ float fast_rcp(float x) { return __builtin_amdgcn_rcpf(x); }

// ---- WMMA fragment helpers (16x16x32 f16, wave32 layouts per CDNA5 ISA 7.12.2) ----

// A-frag: lane l holds row m=l&15; halves 0..7 = K (l>>4)*8 + j, halves 8..15 = K 16.. (zero, K=16)
__device__ __forceinline__ v16h ld_afrag(_Float16 (*tile)[16], int lane) {
  int m = lane & 15, hi = (lane >> 4) & 1;
  v16h a = {};
  #pragma unroll
  for (int j = 0; j < 8; ++j) a[j] = tile[m][hi * 8 + j];
  return a;
}

// B-frag from pre-packed per-lane storage: one 16B load (8 f16), upper K half zero
__device__ __forceinline__ v16h ld_bpacked(const _Float16* __restrict__ P, int idx) {
  v8h lo = *(const v8h*)(P + (size_t)idx * 8);
  v16h b = {};
  #pragma unroll
  for (int j = 0; j < 8; ++j) b[j] = lo[j];
  return b;
}

// C-frag: lane l holds n=l&15, VGPR r -> row m = r + 8*(l>>4). Store silu(c*scale) as f16 tile.
__device__ __forceinline__ void st_h_silu(_Float16 (*tile)[16], int lane, v8f c, float scale) {
  int n = lane & 15, hi = (lane >> 4) & 1;
  #pragma unroll
  for (int r = 0; r < 8; ++r) {
    float h = c[r] * scale;
    h = h * fast_rcp(1.0f + __expf(-h));   // silu via v_rcp_f32
    tile[r + 8 * hi][n] = (_Float16)h;
  }
}

__device__ __forceinline__ void st_tile(float (*buf)[17], int lane, int mbase, v8f c) {
  int n = lane & 15, hi = (lane >> 4) & 1;
  #pragma unroll
  for (int r = 0; r < 8; ++r) buf[mbase + r + 8 * hi][n] = c[r] * 0.25f;  // /sqrt(16)
}

// ---------------- Pass 0: pack B fragments (edge-independent, tiny) ----------------
__global__ void se_pack_kernel(const float* __restrict__ Wk1,
                               const float* __restrict__ Wk2,
                               const float* __restrict__ Wv1,
                               const float* __restrict__ Wv2,
                               _Float16* __restrict__ bk1P,   // [32][8]
                               _Float16* __restrict__ bv1P,   // [32][8]
                               _Float16* __restrict__ bk2P,   // [36][32][8]
                               _Float16* __restrict__ bv2P) { // [36][32][8]
  int tid = blockIdx.x * blockDim.x + threadIdx.x;
  if (tid >= NT2 * 32) return;
  int lane = tid & 31;
  int t = tid >> 5;
  int n = lane & 15, hi = (lane >> 4) & 1;
  #pragma unroll
  for (int j = 0; j < 8; ++j) {
    bk2P[(size_t)tid * 8 + j] = (_Float16)Wk2[(hi * 8 + j) * WNUMC + 16 * t + n];
    bv2P[(size_t)tid * 8 + j] = (_Float16)Wv2[(hi * 8 + j) * WNUMC + 16 * t + n];
  }
  if (t == 0) {
    #pragma unroll
    for (int j = 0; j < 8; ++j) {
      bk1P[(size_t)lane * 8 + j] = (_Float16)Wk1[(hi * 8 + j) * 16 + n];
      bv1P[(size_t)lane * 8 + j] = (_Float16)Wv1[(hi * 8 + j) * 16 + n];
    }
  }
}

// ---------------- Pass 1: per-node q0/q1 ----------------
__global__ void se_q_kernel(const float* __restrict__ x,
                            const float* __restrict__ Wq0,
                            const float* __restrict__ Wq1,
                            float* __restrict__ q0W,
                            float* __restrict__ q1W, int N) {
  int n = blockIdx.x * blockDim.x + threadIdx.x;
  if (n >= N) return;
  const float* xr = x + (size_t)n * DCH;
  float xs0[16];
  #pragma unroll
  for (int u = 0; u < 16; ++u) xs0[u] = xr[u];
  #pragma unroll
  for (int w = 0; w < 16; ++w) {
    float s = 0.f;
    #pragma unroll
    for (int u = 0; u < 16; ++u) s += xs0[u] * Wq0[u * 16 + w];
    q0W[(size_t)n * 16 + w] = s * 0.25f;                        // /sqrt(16)
  }
  float x1[24];
  #pragma unroll
  for (int j = 0; j < 24; ++j) x1[j] = xr[16 + j];
  const float r8 = 0.35355339059f;                               // 1/sqrt(8)
  #pragma unroll
  for (int w = 0; w < 8; ++w)
    #pragma unroll
    for (int i = 0; i < 3; ++i) {
      float s = 0.f;
      #pragma unroll
      for (int u = 0; u < 8; ++u) s += x1[u * 3 + i] * Wq1[u * 8 + w];
      q1W[(size_t)n * 24 + w * 3 + i] = s * r8;
    }
}

// ---------------- Pass 2: fused edge kernel (1 wave = 32 edges) ----------------
__global__ __launch_bounds__(32)
void se_edge_kernel(const float* __restrict__ x,
                    const int*   __restrict__ ei,
                    const float* __restrict__ eattr,
                    const float* __restrict__ amf,
                    const _Float16* __restrict__ bk1P,
                    const _Float16* __restrict__ bv1P,
                    const _Float16* __restrict__ bk2P,
                    const _Float16* __restrict__ bv2P,
                    const float* __restrict__ Wd0,
                    const float* __restrict__ Wd1,
                    const float* __restrict__ q0W,
                    const float* __restrict__ q1W,
                    float* __restrict__ zW,
                    float* __restrict__ expvW,
                    float* __restrict__ vbuf,
                    int E) {
  __shared__ _Float16 ldsT[2][16][16];   // emb / h staging tiles (reused)
  __shared__ float wkt[32][17];          // current wk N-tile (padded, conflict-free)
  __shared__ float wvt[32][17];
  __shared__ float xs0L[32][17];
  __shared__ float xs1L[32][25];
  __shared__ float dotL[32][9];
  __shared__ float q0L[32][17];
  __shared__ float q1L[32][25];
  __shared__ float Wd0L[256];
  __shared__ float Wd1L[64];

  const int lane = threadIdx.x;
  int eg = blockIdx.x * 32 + lane;
  const bool valid = eg < E;
  const int ec = valid ? eg : (E - 1);

  const int src = ei[ec];
  const int dst = ei[E + ec];
  const float inv_sqrt3 = 0.57735026919f;

  // gather per-edge inputs into per-lane LDS rows (needed later with runtime u index)
  const float* xr = x + (size_t)src * DCH;
  #pragma unroll
  for (int u = 0; u < 16; ++u) xs0L[lane][u] = xr[u];

  const float sh0 = eattr[(size_t)ec * 4 + 0];
  float sh1[3];
  sh1[0] = eattr[(size_t)ec * 4 + 1];
  sh1[1] = eattr[(size_t)ec * 4 + 2];
  sh1[2] = eattr[(size_t)ec * 4 + 3];

  #pragma unroll
  for (int u = 0; u < 8; ++u) {
    float a = xr[16 + u * 3 + 0];
    float b = xr[16 + u * 3 + 1];
    float c = xr[16 + u * 3 + 2];
    xs1L[lane][u * 3 + 0] = a;
    xs1L[lane][u * 3 + 1] = b;
    xs1L[lane][u * 3 + 2] = c;
    dotL[lane][u] = (a * sh1[0] + b * sh1[1] + c * sh1[2]) * inv_sqrt3;
  }

  const float* q0p = q0W + (size_t)dst * 16;
  #pragma unroll
  for (int u = 0; u < 16; ++u) q0L[lane][u] = q0p[u];
  const float* q1p = q1W + (size_t)dst * 24;
  #pragma unroll
  for (int u = 0; u < 24; ++u) q1L[lane][u] = q1p[u];

  // radial embedding: emb_j = C * sus(ud - j) * sus(j + 2 - ud),  ud = d*17/8
  const float d  = amf[ec];
  const float ud = d * 2.125f;
  #pragma unroll
  for (int j = 0; j < 16; ++j) {
    float t1 = ud - (float)j;
    float t2 = (float)(j + 2) - ud;
    float s1 = (t1 > 0.f) ? __expf(-fast_rcp(t1)) : 0.0f;
    float s2 = (t2 > 0.f) ? __expf(-fast_rcp(t2)) : 0.0f;
    ldsT[lane >> 4][lane & 15][j] = (_Float16)(33.7343006f * s1 * s2);  // 1.14136*e^2*sqrt(16)
  }
  float tc = 10.0f * (1.0f - d * 0.125f);
  float cutoff = (tc > 0.f) ? __expf(-fast_rcp(tc)) : 0.0f;

  for (int idx = lane; idx < 256; idx += 32) Wd0L[idx] = Wd0[idx];
  Wd1L[lane] = Wd1[lane];
  Wd1L[lane + 32] = Wd1[lane + 32];
  __syncthreads();

  // GEMM1: h = silu(emb @ W1 / 4), both M-tiles, k and v paths
  v16h a_e0 = ld_afrag(ldsT[0], lane);
  v16h a_e1 = ld_afrag(ldsT[1], lane);
  v8f  cz   = {};

  v16h bk1 = ld_bpacked(bk1P, lane);
  v8f hk0 = __builtin_amdgcn_wmma_f32_16x16x32_f16(false, a_e0, false, bk1, (short)0, cz, false, false);
  v8f hk1 = __builtin_amdgcn_wmma_f32_16x16x32_f16(false, a_e1, false, bk1, (short)0, cz, false, false);
  __syncthreads();
  st_h_silu(ldsT[0], lane, hk0, 0.25f);
  st_h_silu(ldsT[1], lane, hk1, 0.25f);
  __syncthreads();
  v16h a_hk0 = ld_afrag(ldsT[0], lane);
  v16h a_hk1 = ld_afrag(ldsT[1], lane);

  v16h bv1 = ld_bpacked(bv1P, lane);
  v8f hv0 = __builtin_amdgcn_wmma_f32_16x16x32_f16(false, a_e0, false, bv1, (short)0, cz, false, false);
  v8f hv1 = __builtin_amdgcn_wmma_f32_16x16x32_f16(false, a_e1, false, bv1, (short)0, cz, false, false);
  __syncthreads();
  st_h_silu(ldsT[0], lane, hv0, 0.25f);
  st_h_silu(ldsT[1], lane, hv1, 0.25f);
  __syncthreads();
  v16h a_hv0 = ld_afrag(ldsT[0], lane);
  v16h a_hv1 = ld_afrag(ldsT[1], lane);

  // fctp accumulators (statically indexed registers only)
  float o0k[16], o0v[16];
  #pragma unroll
  for (int j = 0; j < 16; ++j) { o0k[j] = 0.f; o0v[j] = 0.f; }
  float s2k[8], s2v[8];
  #pragma unroll
  for (int j = 0; j < 8; ++j) { s2k[j] = 0.f; s2v[j] = 0.f; }
  float s3k[8][3], s3v[8][3];
  #pragma unroll
  for (int j = 0; j < 8; ++j)
    #pragma unroll
    for (int i = 0; i < 3; ++i) { s3k[j][i] = 0.f; s3v[j][i] = 0.f; }

  // GEMM2 tile producer: wk/wv channels [16t, 16t+16) for all 32 edges
  auto make_tile = [&](int t) {
    v16h bk = ld_bpacked(bk2P, t * 32 + lane);
    v16h bv = ld_bpacked(bv2P, t * 32 + lane);
    v8f ck0 = __builtin_amdgcn_wmma_f32_16x16x32_f16(false, a_hk0, false, bk, (short)0, cz, false, false);
    v8f ck1 = __builtin_amdgcn_wmma_f32_16x16x32_f16(false, a_hk1, false, bk, (short)0, cz, false, false);
    v8f cv0 = __builtin_amdgcn_wmma_f32_16x16x32_f16(false, a_hv0, false, bv, (short)0, cz, false, false);
    v8f cv1 = __builtin_amdgcn_wmma_f32_16x16x32_f16(false, a_hv1, false, bv, (short)0, cz, false, false);
    __syncthreads();
    st_tile(wkt, lane, 0, ck0);  st_tile(wkt, lane, 16, ck1);
    st_tile(wvt, lane, 0, cv0);  st_tile(wvt, lane, 16, cv1);
    __syncthreads();
  };

  // group 1: channels [0,256) = w1[u=t][w=j] : o0 += xs0[u]*sh0 * w1[u][w]
  for (int t = 0; t < 16; ++t) {
    make_tile(t);
    float c0 = sh0 * xs0L[lane][t];
    #pragma unroll
    for (int j = 0; j < 16; ++j) { o0k[j] += c0 * wkt[lane][j]; o0v[j] += c0 * wvt[lane][j]; }
  }
  // group 2: channels [256,384) = w2[u][w], u = 2t-32 + (j>>3), w = j&7
  for (int t = 16; t < 24; ++t) {
    make_tile(t);
    int u0 = 2 * t - 32;
    float ca = xs0L[lane][u0], cb = xs0L[lane][u0 + 1];
    #pragma unroll
    for (int j = 0; j < 8; ++j) {
      s2k[j] += ca * wkt[lane][j] + cb * wkt[lane][8 + j];
      s2v[j] += ca * wvt[lane][j] + cb * wvt[lane][8 + j];
    }
  }
  // group 3: channels [384,448) = w3[u][w]: s3[w][i] += xs1[u][i] * w3[u][w]
  for (int t = 24; t < 28; ++t) {
    make_tile(t);
    int u0 = 2 * t - 48;
    float xa0 = xs1L[lane][u0 * 3 + 0], xa1 = xs1L[lane][u0 * 3 + 1], xa2 = xs1L[lane][u0 * 3 + 2];
    float xb0 = xs1L[lane][u0 * 3 + 3], xb1 = xs1L[lane][u0 * 3 + 4], xb2 = xs1L[lane][u0 * 3 + 5];
    #pragma unroll
    for (int j = 0; j < 8; ++j) {
      float wka = wkt[lane][j], wkb = wkt[lane][8 + j];
      float wva = wvt[lane][j], wvb = wvt[lane][8 + j];
      s3k[j][0] += xa0 * wka + xb0 * wkb;
      s3k[j][1] += xa1 * wka + xb1 * wkb;
      s3k[j][2] += xa2 * wka + xb2 * wkb;
      s3v[j][0] += xa0 * wva + xb0 * wvb;
      s3v[j][1] += xa1 * wva + xb1 * wvb;
      s3v[j][2] += xa2 * wva + xb2 * wvb;
    }
  }
  // group 4: channels [448,576) = w4[u=t-28][w=j] : o0 += dot11[u] * w4[u][w]
  for (int t = 28; t < 36; ++t) {
    make_tile(t);
    float cd = dotL[lane][t - 28];
    #pragma unroll
    for (int j = 0; j < 16; ++j) { o0k[j] += cd * wkt[lane][j]; o0v[j] += cd * wvt[lane][j]; }
  }

  const float norm = 0.20412414523f;      // 1/sqrt(M0+M1)
  float k0f[16], v0f[16];
  #pragma unroll
  for (int j = 0; j < 16; ++j) { k0f[j] = o0k[j] * norm; v0f[j] = o0v[j] * norm; }
  float o1k[8][3], o1v[8][3];
  #pragma unroll
  for (int w = 0; w < 8; ++w)
    #pragma unroll
    for (int i = 0; i < 3; ++i) {
      o1k[w][i] = (s2k[w] * sh1[i] + sh0 * s3k[w][i]) * norm;
      o1v[w][i] = (s2v[w] * sh1[i] + sh0 * s3v[w][i]) * norm;
    }

  // score = (q0·Wd0·k0 + q1·Wd1·k1/sqrt3) / sqrt(320)
  float acc0[16];
  #pragma unroll
  for (int j = 0; j < 16; ++j) acc0[j] = 0.f;
  for (int u = 0; u < 16; ++u) {
    float qa = q0L[lane][u];
    #pragma unroll
    for (int vv = 0; vv < 16; ++vv) acc0[vv] += qa * Wd0L[u * 16 + vv];
  }
  float score0 = 0.f;
  #pragma unroll
  for (int vv = 0; vv < 16; ++vv) score0 += acc0[vv] * k0f[vv];

  float acc1[8][3];
  #pragma unroll
  for (int w = 0; w < 8; ++w)
    #pragma unroll
    for (int i = 0; i < 3; ++i) acc1[w][i] = 0.f;
  for (int u = 0; u < 8; ++u) {
    float qa0 = q1L[lane][u * 3 + 0];
    float qa1 = q1L[lane][u * 3 + 1];
    float qa2 = q1L[lane][u * 3 + 2];
    #pragma unroll
    for (int vv = 0; vv < 8; ++vv) {
      float wd = Wd1L[u * 8 + vv];
      acc1[vv][0] += qa0 * wd;
      acc1[vv][1] += qa1 * wd;
      acc1[vv][2] += qa2 * wd;
    }
  }
  float score1 = 0.f;
  #pragma unroll
  for (int vv = 0; vv < 8; ++vv)
    #pragma unroll
    for (int i = 0; i < 3; ++i) score1 += acc1[vv][i] * o1k[vv][i];

  float score = (score0 + score1 * inv_sqrt3) * 0.05590169944f;  // 1/sqrt(320)
  float expv  = cutoff * __expf(score);

  if (valid) {
    expvW[eg] = expv;
    atomicAdd(&zW[dst], expv);
    float* vo = vbuf + (size_t)eg * DCH;
    #pragma unroll
    for (int j = 0; j < 16; ++j) vo[j] = v0f[j];
    #pragma unroll
    for (int w = 0; w < 8; ++w)
      #pragma unroll
      for (int i = 0; i < 3; ++i) vo[16 + w * 3 + i] = o1v[w][i];
  }
}

// ---------------- Pass 3: normalize + scatter ----------------
__global__ void se_scatter_kernel(const int* __restrict__ ei,
                                  const float* __restrict__ expvW,
                                  const float* __restrict__ zW,
                                  const float* __restrict__ vbuf,
                                  float* __restrict__ out, int E) {
  int e = blockIdx.x * blockDim.x + threadIdx.x;
  if (e >= E) return;
  int dst = ei[E + e];
  float zz = zW[dst];
  if (zz == 0.0f) zz = 1.0f;
  float alpha = expvW[e] * __builtin_amdgcn_rcpf(zz);
  float we = sqrtf(fmaxf(alpha, 0.0f));
  const float* ve = vbuf + (size_t)e * DCH;
  float* od = out + (size_t)dst * DCH;
  #pragma unroll
  for (int c = 0; c < DCH; ++c) atomicAdd(&od[c], we * ve[c]);
}

extern "C" void kernel_launch(void* const* d_in, const int* in_sizes, int n_in,
                              void* d_out, int out_size, void* d_ws, size_t ws_size,
                              hipStream_t stream) {
  const float* x     = (const float*)d_in[0];
  const int*   ei    = (const int*)  d_in[1];
  const float* eattr = (const float*)d_in[2];
  // d_in[3] node_attr, d_in[4] batch: unused by the reference math
  const float* amf   = (const float*)d_in[5];
  const float* Wq0   = (const float*)d_in[6];
  const float* Wq1   = (const float*)d_in[7];
  const float* Wk1   = (const float*)d_in[8];
  const float* Wk2   = (const float*)d_in[9];
  const float* Wv1   = (const float*)d_in[10];
  const float* Wv2   = (const float*)d_in[11];
  const float* Wd0   = (const float*)d_in[12];
  const float* Wd1   = (const float*)d_in[13];

  const int N = in_sizes[0] / DCH;
  const int E = in_sizes[1] / 2;

  // workspace layout:
  //   f16 packed B-fragments first (16B-aligned):
  //     bk1P[32*8] bv1P[32*8] bk2P[36*32*8] bv2P[36*32*8]   (= 18944 halves, 37888 B)
  //   then f32: q0[N*16] q1[N*24] z[N] expv[E] vbuf[E*40]
  _Float16* bk1P = (_Float16*)d_ws;
  _Float16* bv1P = bk1P + 32 * 8;
  _Float16* bk2P = bv1P + 32 * 8;
  _Float16* bv2P = bk2P + NT2 * 32 * 8;
  float* fws   = (float*)(bv2P + NT2 * 32 * 8);
  float* q0W   = fws;
  float* q1W   = q0W + (size_t)N * 16;
  float* zW    = q1W + (size_t)N * 24;
  float* expvW = zW + (size_t)N;
  float* vbuf  = expvW + (size_t)E;

  hipMemsetAsync(zW, 0, sizeof(float) * (size_t)N, stream);
  hipMemsetAsync(d_out, 0, sizeof(float) * (size_t)out_size, stream);

  se_pack_kernel<<<(NT2 * 32 + 255) / 256, 256, 0, stream>>>(
      Wk1, Wk2, Wv1, Wv2, bk1P, bv1P, bk2P, bv2P);

  se_q_kernel<<<(N + 255) / 256, 256, 0, stream>>>(x, Wq0, Wq1, q0W, q1W, N);

  se_edge_kernel<<<(E + 31) / 32, 32, 0, stream>>>(
      x, ei, eattr, amf, bk1P, bv1P, bk2P, bv2P, Wd0, Wd1,
      q0W, q1W, zW, expvW, vbuf, E);

  se_scatter_kernel<<<(E + 255) / 256, 256, 0, stream>>>(
      ei, expvW, zW, vbuf, (float*)d_out, E);
}
